// GAT_61503931678814
// MI455X (gfx1250) — matrix operand order repaired and checked by reading, not compiled, not run on hardware
//
#include <hip/hip_runtime.h>
#include <hip/hip_bf16.h>
#include <math.h>

// ---- problem constants (match reference) ----
#define NN   20000   // nodes
#define EE   40000   // edges
#define GG   1000    // graphs
#define FF   64      // node feature dim
#define EFD  16      // edge feature dim
#define HH   8       // heads
#define DD   128     // head dim
#define HD   1024    // H*D
#define CC   17      // EF + 1 (bias channel)
#define TC   1088    // CC * FF  (columns of T)

typedef __attribute__((ext_vector_type(16))) _Float16 v16h;
typedef __attribute__((ext_vector_type(8)))  _Float16 v8h;
typedef __attribute__((ext_vector_type(8)))  float    v8f;

// ---------------------------------------------------------------------------
// small helpers
// ---------------------------------------------------------------------------
__device__ __forceinline__ float warp_red32(float v) {
  #pragma unroll
  for (int off = 16; off > 0; off >>= 1) v += __shfl_down(v, off, 32);
  return v;
}

__device__ __forceinline__ void atomicMaxF(float* addr, float val) {
  unsigned int* ai = (unsigned int*)addr;
  unsigned int cur = *ai;
  while (__uint_as_float(cur) < val) {
    unsigned int assumed = cur;
    cur = atomicCAS(ai, assumed, __float_as_uint(val));
    if (cur == assumed) break;
  }
}

// ---------------------------------------------------------------------------
// elementwise utilities
// ---------------------------------------------------------------------------
__global__ void k_fill(float* p, float v, int n) {
  for (int i = blockIdx.x * blockDim.x + threadIdx.x; i < n; i += gridDim.x * blockDim.x)
    p[i] = v;
}

__global__ void k_f32_to_f16(const float* __restrict__ x, _Float16* __restrict__ y, int n) {
  for (int i = blockIdx.x * blockDim.x + threadIdx.x; i < n; i += gridDim.x * blockDim.x)
    y[i] = (_Float16)x[i];
}

// out[n*64+f] = bias[f]  (init message accumulator with nn bias)
__global__ void k_init_bias(float* __restrict__ out, const float* __restrict__ bias, int n) {
  for (int i = blockIdx.x * blockDim.x + threadIdx.x; i < n; i += gridDim.x * blockDim.x)
    out[i] = bias[i & (FF - 1)];
}

// ---------------------------------------------------------------------------
// Build the concatenated NNConv weight, transposed + f16:
//   wt[(c*64+g)*64 + f] = (c<16) ? w[c*4096 + f*64 + g] : b[f*64 + g]
// grid = TC blocks, block = 64 threads (f)
// ---------------------------------------------------------------------------
__global__ void k_build_wcat(const float* __restrict__ w, const float* __restrict__ b,
                             _Float16* __restrict__ wt) {
  int col = blockIdx.x;          // 0..1087
  int f   = threadIdx.x;         // 0..63
  int c = col >> 6, g = col & 63;
  float v = (c < EFD) ? w[c * (FF * FF) + f * FF + g] : b[f * FF + g];
  wt[col * FF + f] = (_Float16)v;
}

// Transpose f32 [K x Ncols] -> f16 [Ncols x K] (for gat_w: K=64, Ncols=1024)
__global__ void k_transpose_to_f16(const float* __restrict__ w, _Float16* __restrict__ wt,
                                   int K, int Ncols) {
  int col = blockIdx.x;          // 0..Ncols-1
  int k   = threadIdx.x;         // 0..K-1
  wt[col * K + k] = (_Float16)w[k * Ncols + col];
}

// ---------------------------------------------------------------------------
// WMMA GEMM:  C[M x KOUT] (f32) = A[M x 64] (f16, row-major) @ B
// B supplied pre-transposed: Bt[KOUT x 64] f16 so each lane's fragment is a
// pair of 8-element runs along K (k = 8*khalf + j and 16 + 8*khalf + j).
//
// One wave per block (32 threads); wave computes a 16x64 output strip:
//  - A fragments for both K=32 steps loaded once (reused across 4 col tiles)
//  - 4 accumulators, 8 WMMAs per wave
// grid = (M/16, KOUT/64); M%16==0, KOUT%64==0 so EXEC is all-ones at each WMMA.
// KOUT is a template constant so the row-strided stores use immediate offsets.
// ---------------------------------------------------------------------------
template <int KOUT>
__global__ __launch_bounds__(32)
void k_gemm_wmma(const _Float16* __restrict__ A, const _Float16* __restrict__ Bt,
                 float* __restrict__ C) {
  const int lane  = threadIdx.x & 31;
  const int mt    = blockIdx.x;          // row tile (16 rows)
  const int ntb   = blockIdx.y * 4;      // first of 4 col tiles (64 cols)
  const int row   = mt * 16 + (lane & 15);
  const int khalf = lane >> 4;           // 0 or 1

  // A fragments (16x32 f16 each) for both K-steps, loaded once.
  v16h af[2];
  #pragma unroll
  for (int s = 0; s < 2; ++s) {
    v8h a0 = *(const v8h*)(A + row * FF + s * 32 + khalf * 8);
    v8h a1 = *(const v8h*)(A + row * FF + s * 32 + 16 + khalf * 8);
    #pragma unroll
    for (int j = 0; j < 8; ++j) { af[s][j] = a0[j]; af[s][8 + j] = a1[j]; }
  }

  #pragma unroll
  for (int t = 0; t < 4; ++t) {
    const int col = (ntb + t) * 16 + (lane & 15);
    v8f acc = {};
    #pragma unroll
    for (int s = 0; s < 2; ++s) {
      v8h b0 = *(const v8h*)(Bt + col * FF + s * 32 + khalf * 8);
      v8h b1 = *(const v8h*)(Bt + col * FF + s * 32 + 16 + khalf * 8);
      v16h bf;
      #pragma unroll
      for (int j = 0; j < 8; ++j) { bf[j] = b0[j]; bf[8 + j] = b1[j]; }
      acc = __builtin_amdgcn_wmma_f32_16x16x32_f16(
          /*neg_a=*/false, af[s], /*neg_b=*/false, bf,
          /*c_mod=*/(short)0, acc, /*reuse_a=*/false, /*reuse_b=*/false);
    }
    // C/D layout: acc[v] -> row = mt*16 + khalf*8 + v, col
    float* cp = C + (size_t)(mt * 16 + khalf * 8) * KOUT + col;
    #pragma unroll
    for (int v = 0; v < 8; ++v)
      cp[(size_t)v * KOUT] = acc[v];
  }
}

// ---------------------------------------------------------------------------
// Per-edge NNConv message + scatter:
//   m[e,g] = T[src, 16*64+g] + sum_c e[e,c] * T[src, c*64+g]; atomicAdd to out[dst,g]
// one wave per edge (block 256 = 8 edges), each lane handles 2 features
// ---------------------------------------------------------------------------
__global__ void k_edge_message(const float* __restrict__ eF, const int* __restrict__ src,
                               const int* __restrict__ dst, const float* __restrict__ T,
                               float* __restrict__ out) {
  int e = blockIdx.x * 8 + (threadIdx.x >> 5);
  if (e >= EE) return;
  int lane = threadIdx.x & 31;
  int s = src[e], d = dst[e];
  const float* trow = T + (size_t)s * TC;
  __builtin_prefetch(trow, 0, 1);                 // global_prefetch_b8
  __builtin_prefetch(trow + 544, 0, 1);
  int f0 = lane, f1 = lane + 32;
  float a0 = trow[EFD * FF + f0];                 // bias block (channel weight 1)
  float a1 = trow[EFD * FF + f1];
  #pragma unroll
  for (int c = 0; c < EFD; ++c) {
    float w = eF[e * EFD + c];
    a0 += w * trow[c * FF + f0];
    a1 += w * trow[c * FF + f1];
  }
  atomicAdd(out + (size_t)d * FF + f0, a0);
  atomicAdd(out + (size_t)d * FF + f1, a1);
}

// ---------------------------------------------------------------------------
// BatchNorm (training-mode batch stats) + ReLU
// ---------------------------------------------------------------------------
__global__ void k_bn_stats(const float* __restrict__ x, float* __restrict__ mu,
                           float* __restrict__ var) {
  __shared__ float ss[256], sq[256];
  int f = blockIdx.x, t = threadIdx.x;
  float s = 0.f, q = 0.f;
  for (int n = t; n < NN; n += 256) {
    float v = x[(size_t)n * FF + f];
    s += v; q += v * v;
  }
  ss[t] = s; sq[t] = q; __syncthreads();
  for (int st = 128; st > 0; st >>= 1) {
    if (t < st) { ss[t] += ss[t + st]; sq[t] += sq[t + st]; }
    __syncthreads();
  }
  if (t == 0) {
    float m = ss[0] / (float)NN;
    mu[f] = m;
    var[f] = sq[0] / (float)NN - m * m;
  }
}

__global__ void k_bn_relu(float* __restrict__ x, const float* __restrict__ mu,
                          const float* __restrict__ var, const float* __restrict__ g,
                          const float* __restrict__ b, int n) {
  for (int i = blockIdx.x * blockDim.x + threadIdx.x; i < n; i += gridDim.x * blockDim.x) {
    int f = i & (FF - 1);
    float v = (x[i] - mu[f]) * rsqrtf(var[f] + 1e-5f) * g[f] + b[f];
    x[i] = v > 0.f ? v : 0.f;
  }
}

// ---------------------------------------------------------------------------
// GAT attention pieces
// ---------------------------------------------------------------------------
// el[n,h] = sum_d z[n,h,d]*al[h,d]; er analogous.  grid=N, block=256 (wave==head)
__global__ void k_el_er(const float* __restrict__ z, const float* __restrict__ al,
                        const float* __restrict__ ar, float* __restrict__ el,
                        float* __restrict__ er) {
  int n = blockIdx.x;
  int h = threadIdx.x >> 5, lane = threadIdx.x & 31;
  const float* zr = z + (size_t)n * HD + h * DD;
  const float* alr = al + h * DD;
  const float* arr = ar + h * DD;
  float sl = 0.f, sr = 0.f;
  #pragma unroll
  for (int j = 0; j < 4; ++j) {
    int d = lane + 32 * j;
    float zv = zr[d];
    sl += zv * alr[d];
    sr += zv * arr[d];
  }
  sl = warp_red32(sl); sr = warp_red32(sr);
  if (lane == 0) { el[n * HH + h] = sl; er[n * HH + h] = sr; }
}

// logits + segment max.  thread per (edge, head)
__global__ void k_edge_logits_max(const float* __restrict__ el, const float* __restrict__ er,
                                  const int* __restrict__ src, const int* __restrict__ dst,
                                  float* __restrict__ logits, float* __restrict__ gmax) {
  int i = blockIdx.x * blockDim.x + threadIdx.x;
  if (i >= EE * HH) return;
  int e = i >> 3, h = i & 7;
  float l = el[src[e] * HH + h] + er[dst[e] * HH + h];
  l = l > 0.f ? l : 0.2f * l;                 // leaky_relu 0.2
  logits[i] = l;
  atomicMaxF(gmax + dst[e] * HH + h, l);
}

__global__ void k_edge_exp_den(const float* __restrict__ logits, const int* __restrict__ dst,
                               const float* __restrict__ gmax, float* __restrict__ ex,
                               float* __restrict__ den) {
  int i = blockIdx.x * blockDim.x + threadIdx.x;
  if (i >= EE * HH) return;
  int e = i >> 3, h = i & 7;
  float v = expf(logits[i] - gmax[dst[e] * HH + h]);
  ex[i] = v;
  atomicAdd(den + dst[e] * HH + h, v);
}

// rst[dst,h,d] += alpha * z[src,h,d].  block per edge, wave per head, 4 dims/lane
__global__ void k_edge_alpha_accum(const float* __restrict__ ex, const float* __restrict__ den,
                                   const int* __restrict__ src, const int* __restrict__ dst,
                                   const float* __restrict__ z, float* __restrict__ rst) {
  int e = blockIdx.x;
  int h = threadIdx.x >> 5, lane = threadIdx.x & 31;
  float dn = den[dst[e] * HH + h];
  float alpha = ex[e * HH + h] / (dn > 0.f ? dn : 1.f);
  const float* zr = z + (size_t)src[e] * HD + h * DD;
  float* rr = rst + (size_t)dst[e] * HD + h * DD;
  #pragma unroll
  for (int j = 0; j < 4; ++j) {
    int d = lane + 32 * j;
    atomicAdd(rr + d, alpha * zr[d]);
  }
}

// h2 = elu(rst + gat_b)
__global__ void k_elu_bias(float* __restrict__ rst, const float* __restrict__ gb, int n) {
  for (int i = blockIdx.x * blockDim.x + threadIdx.x; i < n; i += gridDim.x * blockDim.x) {
    float v = rst[i] + gb[i & (HD - 1)];
    rst[i] = v > 0.f ? v : expm1f(v);
  }
}

// ---------------------------------------------------------------------------
// Global attention pooling (graphs are contiguous 20-node blocks)
// ---------------------------------------------------------------------------
__global__ void k_gate(const float* __restrict__ h2, const float* __restrict__ gw,
                       const float* __restrict__ gb, float* __restrict__ gate) {
  int n = blockIdx.x * 8 + (threadIdx.x >> 5);
  if (n >= NN) return;
  int lane = threadIdx.x & 31;
  const float* hr = h2 + (size_t)n * HD;
  float s = 0.f;
  for (int k = lane; k < HD; k += 32) s += hr[k] * gw[k];
  s = warp_red32(s);
  if (lane == 0) gate[n] = s + gb[0];
}

__global__ void k_graph_pool(const float* __restrict__ gate, const float* __restrict__ h2,
                             float* __restrict__ r) {
  __shared__ float sa[32];
  __shared__ float smax, ssum;
  const int NPG = NN / GG;                 // 20 nodes per graph
  int g = blockIdx.x, t = threadIdx.x;
  int n0 = g * NPG;
  if (t == 0) {
    float m = -3.0e38f;
    for (int i = 0; i < NPG; ++i) m = fmaxf(m, gate[n0 + i]);
    smax = m;
  }
  __syncthreads();
  if (t < NPG) sa[t] = expf(gate[n0 + t] - smax);
  __syncthreads();
  if (t == 0) {
    float s = 0.f;
    for (int i = 0; i < NPG; ++i) s += sa[i];
    ssum = s;
  }
  __syncthreads();
  for (int dim = t; dim < HD; dim += blockDim.x) {
    float acc = 0.f;
    for (int i = 0; i < NPG; ++i)
      acc += sa[i] * h2[(size_t)(n0 + i) * HD + dim];
    r[(size_t)g * HD + dim] = acc / ssum;
  }
}

// ---------------------------------------------------------------------------
// MLP head
// ---------------------------------------------------------------------------
__global__ void k_mlp1(const float* __restrict__ r, const float* __restrict__ w,
                       const float* __restrict__ b, float* __restrict__ x1) {
  int g = blockIdx.x, t = threadIdx.x;          // block 64
  const float* rr = r + (size_t)g * HD;
  float s = 0.f;
  for (int k = 0; k < HD; ++k) s += rr[k] * w[k * 64 + t];
  s += b[t];
  x1[g * 64 + t] = s > 0.f ? s : 0.f;
}

__global__ void k_mlp2(const float* __restrict__ x1, const float* __restrict__ w,
                       const float* __restrict__ b, float* __restrict__ x2) {
  int g = blockIdx.x, t = threadIdx.x;          // block 32
  const float* xr = x1 + g * 64;
  float s = 0.f;
  for (int k = 0; k < 64; ++k) s += xr[k] * w[k * 32 + t];
  s += b[t];
  x2[g * 32 + t] = s > 0.f ? s : 0.f;
}

__global__ void k_mlp3(const float* __restrict__ x2, const float* __restrict__ w,
                       const float* __restrict__ b, float* __restrict__ out) {
  int g = blockIdx.x;                            // block 32
  int lane = threadIdx.x & 31;
  float v = x2[g * 32 + lane] * w[lane];
  v = warp_red32(v);
  if (lane == 0) out[g] = v + b[0];
}

// ---------------------------------------------------------------------------
// host orchestration
// ---------------------------------------------------------------------------
static inline size_t alignup(size_t x) { return (x + 255) & ~(size_t)255; }

extern "C" void kernel_launch(void* const* d_in, const int* in_sizes, int n_in,
                              void* d_out, int out_size, void* d_ws, size_t ws_size,
                              hipStream_t stream) {
  const float* n_in0  = (const float*)d_in[0];
  const float* eF     = (const float*)d_in[1];
  const int*   src    = (const int*)d_in[2];
  const int*   dst    = (const int*)d_in[3];
  // d_in[4] = graph_ids (contiguous blocks; unused directly)

  // workspace carve-up
  char* ws = (char*)d_ws;
  size_t off = 0;
  auto take = [&](size_t bytes) { char* p = ws + off; off += alignup(bytes); return p; };
  float*    hA    = (float*)   take((size_t)NN * FF * 4);
  float*    hB    = (float*)   take((size_t)NN * FF * 4);
  _Float16* h16   = (_Float16*)take((size_t)NN * FF * 2);
  _Float16* Wt    = (_Float16*)take((size_t)TC * FF * 2);
  float*    T     = (float*)   take((size_t)NN * TC * 4);   // also z
  float*    rst   = (float*)   take((size_t)NN * HD * 4);   // also h2
  float*    mu    = (float*)   take(FF * 4);
  float*    var   = (float*)   take(FF * 4);
  float*    el    = (float*)   take((size_t)NN * HH * 4);
  float*    er    = (float*)   take((size_t)NN * HH * 4);
  float*    logit = (float*)   take((size_t)EE * HH * 4);
  float*    ex    = (float*)   take((size_t)EE * HH * 4);
  float*    gmax  = (float*)   take((size_t)NN * HH * 4);
  float*    den   = (float*)   take((size_t)NN * HH * 4);
  float*    gate  = (float*)   take((size_t)NN * 4);
  float*    r     = (float*)   take((size_t)GG * HD * 4);
  float*    x1    = (float*)   take((size_t)GG * 64 * 4);
  float*    x2    = (float*)   take((size_t)GG * 32 * 4);

  auto ew_grid = [&](size_t n) { return dim3((unsigned)((n + 255) / 256 < 4096 ? (n + 255) / 256 : 4096)); };

  // ---- 4 NNConv + BN + ReLU layers ----
  const float* hcur = n_in0;
  float* hbufs[2] = { hA, hB };
  for (int layer = 0; layer < 4; ++layer) {
    int pb = 5 + layer * 5;   // edgeX_w, edgeX_b, nn_bias, bn_g, bn_b
    const float* wE  = (const float*)d_in[pb + 0];
    const float* bE  = (const float*)d_in[pb + 1];
    const float* nnb = (const float*)d_in[pb + 2];
    const float* bng = (const float*)d_in[pb + 3];
    const float* bnb = (const float*)d_in[pb + 4];
    float* hnext = hbufs[layer & 1];

    k_build_wcat<<<TC, FF, 0, stream>>>(wE, bE, Wt);
    k_f32_to_f16<<<ew_grid((size_t)NN * FF), 256, 0, stream>>>(hcur, h16, NN * FF);
    k_gemm_wmma<TC><<<dim3(NN / 16, TC / 64), 32, 0, stream>>>(h16, Wt, T);
    k_init_bias<<<ew_grid((size_t)NN * FF), 256, 0, stream>>>(hnext, nnb, NN * FF);
    k_edge_message<<<EE / 8, 256, 0, stream>>>(eF, src, dst, T, hnext);
    k_bn_stats<<<FF, 256, 0, stream>>>(hnext, mu, var);
    k_bn_relu<<<ew_grid((size_t)NN * FF), 256, 0, stream>>>(hnext, mu, var, bng, bnb, NN * FF);
    hcur = hnext;
  }

  // ---- GAT ----
  const float* gatw = (const float*)d_in[25];
  const float* al   = (const float*)d_in[26];
  const float* ar   = (const float*)d_in[27];
  const float* gatb = (const float*)d_in[28];

  k_transpose_to_f16<<<HD, FF, 0, stream>>>(gatw, Wt, FF, HD);
  k_f32_to_f16<<<ew_grid((size_t)NN * FF), 256, 0, stream>>>(hcur, h16, NN * FF);
  k_gemm_wmma<HD><<<dim3(NN / 16, HD / 64), 32, 0, stream>>>(h16, Wt, T);  // T = z
  k_el_er<<<NN, 256, 0, stream>>>(T, al, ar, el, er);

  k_fill<<<ew_grid((size_t)NN * HH), 256, 0, stream>>>(gmax, -3.0e38f, NN * HH);
  k_fill<<<ew_grid((size_t)NN * HH), 256, 0, stream>>>(den, 0.f, NN * HH);
  k_fill<<<ew_grid((size_t)NN * HD), 256, 0, stream>>>(rst, 0.f, NN * HD);

  k_edge_logits_max<<<(EE * HH + 255) / 256, 256, 0, stream>>>(el, er, src, dst, logit, gmax);
  k_edge_exp_den<<<(EE * HH + 255) / 256, 256, 0, stream>>>(logit, dst, gmax, ex, den);
  k_edge_alpha_accum<<<EE, 256, 0, stream>>>(ex, den, src, dst, T, rst);
  k_elu_bias<<<ew_grid((size_t)NN * HD), 256, 0, stream>>>(rst, gatb, NN * HD);  // rst = h2

  // ---- pooling ----
  const float* gw = (const float*)d_in[29];
  const float* gb = (const float*)d_in[30];
  k_gate<<<(NN + 7) / 8, 256, 0, stream>>>(rst, gw, gb, gate);
  k_graph_pool<<<GG, 256, 0, stream>>>(gate, rst, r);

  // ---- MLP head ----
  k_mlp1<<<GG, 64, 0, stream>>>(r, (const float*)d_in[31], (const float*)d_in[32], x1);
  k_mlp2<<<GG, 32, 0, stream>>>(x1, (const float*)d_in[33], (const float*)d_in[34], x2);
  k_mlp3<<<GG, 32, 0, stream>>>(x2, (const float*)d_in[35], (const float*)d_in[36], (float*)d_out);

  (void)in_sizes; (void)n_in; (void)out_size; (void)ws_size;
}
